// SequencePredictorLSTM_14482629722298
// MI455X (gfx1250) — matrix operand order repaired and checked by reading, not compiled, not run on hardware
//
#include <hip/hip_runtime.h>
#include <hip/hip_bf16.h>

typedef __attribute__((ext_vector_type(16))) _Float16 v16h;
typedef __attribute__((ext_vector_type(8)))  _Float16 v8h;
typedef __attribute__((ext_vector_type(8)))  float    v8f;

namespace {
constexpr int Bb = 64;
constexpr int Tt = 2048;
constexpr int Ff = 8;
constexpr int Hh = 128;
constexpr int NG = 4 * Hh;     // 512 gate outputs (i,f,g,o blocks of 128)
constexpr int K0 = 160;        // layer0 concat K: 128 (h) + 8 (x) + 24 zero pad
constexpr int K1 = 256;        // layer1 concat K: 128 (h1_prev) + 128 (h0_t)
constexpr int ROWS = 16;       // batch rows per workgroup (Bb/4)

// LDS strides (in elements), padded for bank-conflict avoidance, 16B-aligned
constexpr int S_HX0 = 168;     // halves per row of hx0 (336 B)
constexpr int S_HX1 = 264;     // halves per row of hx1 (528 B)
constexpr int S_W1  = 264;     // halves per N-row of W1 concat (528 B)
constexpr int S_G   = 516;     // floats per row of gate buffer (2064 B)

constexpr size_t SMEM_W1  = (size_t)NG * S_W1 * 2;    // 270336
constexpr size_t SMEM_HX0 = (size_t)ROWS * S_HX0 * 2; //   5376
constexpr size_t SMEM_HX1 = (size_t)ROWS * S_HX1 * 2; //   8448
constexpr size_t SMEM_G   = (size_t)ROWS * S_G * 4;   //  33024
constexpr size_t SMEM_WL  = (size_t)Ff * Hh * 4;      //   4096
constexpr size_t SMEM_TOTAL = SMEM_W1 + SMEM_HX0 + SMEM_HX1 + SMEM_G + SMEM_WL; // 321280 <= 327680

constexpr float LOG2E = 1.4426950408889634f;
} // namespace

// Fast branchless activations built from v_exp_f32 (2^x) + v_rcp_f32.
//   sigm(x) = rcp(1 + exp2(-log2e*x))
//   tanh(x) = 2*sigm(2x) - 1
__device__ __forceinline__ float fast_sigm_pre(float x_scaled) {
  return __builtin_amdgcn_rcpf(1.0f + __builtin_amdgcn_exp2f(x_scaled));
}
__device__ __forceinline__ float fast_tanh(float x) {
  return __builtin_fmaf(2.0f, fast_sigm_pre(-2.0f * LOG2E * x), -1.0f);
}

__device__ __forceinline__ v16h cat8(v8h lo, v8h hi) {
  return __builtin_shufflevector(lo, hi, 0, 1, 2, 3, 4, 5, 6, 7, 8, 9, 10, 11, 12, 13, 14, 15);
}

__device__ __forceinline__ v8f wmma_f16(v16h a, v16h b, v8f c) {
  return __builtin_amdgcn_wmma_f32_16x16x32_f16(false, a, false, b, (short)0, c, false, false);
}

// ---------------------------------------------------------------------------
// Init kernel: build layer-0 concat weights as f16 [N=512][K0=160] in d_ws.
// ---------------------------------------------------------------------------
__global__ void lstm_w0cat_build(const float* __restrict__ w_hh0,
                                 const float* __restrict__ w_ih0,
                                 _Float16* __restrict__ w0cat) {
  int i = blockIdx.x * blockDim.x + threadIdx.x;
  if (i >= NG * K0) return;
  int n = i / K0, k = i % K0;
  float v = 0.0f;
  if (k < Hh)             v = w_hh0[n * Hh + k];
  else if (k < Hh + Ff)   v = w_ih0[n * Ff + (k - Hh)];
  w0cat[i] = (_Float16)v;
}

// ---------------------------------------------------------------------------
// Fused persistent 2-layer LSTM + linear head. 4 workgroups x 512 threads.
// Each workgroup owns 16 batch rows and scans all T=2048 steps.
// ---------------------------------------------------------------------------
__global__ void __launch_bounds__(512)
lstm_fused_scan(const float* __restrict__ x,
                const float* __restrict__ b_ih0, const float* __restrict__ b_hh0,
                const float* __restrict__ w_hh1, const float* __restrict__ w_ih1,
                const float* __restrict__ b_ih1, const float* __restrict__ b_hh1,
                const float* __restrict__ w_lin, const float* __restrict__ b_lin,
                const _Float16* __restrict__ w0cat,
                float* __restrict__ out) {
  extern __shared__ char smem_raw[];
  _Float16* W1  = (_Float16*)smem_raw;                 // [NG][S_W1]  layer1 concat weights f16
  _Float16* hx0 = (_Float16*)(smem_raw + SMEM_W1);     // [16][S_HX0] {h0_prev(128), x_t(8), pad}
  _Float16* hx1 = (_Float16*)(smem_raw + SMEM_W1 + SMEM_HX0); // [16][S_HX1] {h1_prev(128), h0_t(128)}
  float*    gts = (float*)(smem_raw + SMEM_W1 + SMEM_HX0 + SMEM_HX1); // [16][S_G] activated gates
  float*    wl  = (float*)(smem_raw + SMEM_W1 + SMEM_HX0 + SMEM_HX1 + SMEM_G); // [8][128]

  const int tid   = threadIdx.x;
  const int lane  = tid & 31;
  const int wv    = tid >> 5;     // wave id 0..15
  const int ln    = lane & 15;    // lane % 16
  const int lh    = lane >> 4;    // lane half 0/1
  const int bbase = blockIdx.x * ROWS;

  // --- one-time LDS fill: W1 concat [n][k] = k<128 ? w_hh1[n][k] : w_ih1[n][k-128]
  for (int i = tid; i < NG * K1; i += 512) {
    int n = i >> 8, k = i & 255;
    float v = (k < Hh) ? w_hh1[n * Hh + k] : w_ih1[n * Hh + (k - Hh)];
    W1[n * S_W1 + k] = (_Float16)v;
  }
  for (int i = tid; i < ROWS * S_HX0; i += 512) hx0[i] = (_Float16)0.0f;
  for (int i = tid; i < ROWS * S_HX1; i += 512) hx1[i] = (_Float16)0.0f;
  for (int i = tid; i < Ff * Hh; i += 512) wl[i] = w_lin[i];

  // --- per-wave tile ownership: N-tiles {2*wv, 2*wv+1} of 32 (16 columns each)
  const int tn0 = wv * 2, tn1 = wv * 2 + 1;
  const float bia00 = b_ih0[tn0 * 16 + ln] + b_hh0[tn0 * 16 + ln];
  const float bia01 = b_ih0[tn1 * 16 + ln] + b_hh0[tn1 * 16 + ln];
  const float bia10 = b_ih1[tn0 * 16 + ln] + b_hh1[tn0 * 16 + ln];
  const float bia11 = b_ih1[tn1 * 16 + ln] + b_hh1[tn1 * 16 + ln];

  // Branchless activation constants per tile: act(v) = fma(aout, rcp(1+exp2(ain*v)), aoff)
  const bool t0g = ((tn0 >> 3) == 2), t1g = ((tn1 >> 3) == 2);
  const float ain0 = t0g ? (-2.0f * LOG2E) : (-LOG2E);
  const float ain1 = t1g ? (-2.0f * LOG2E) : (-LOG2E);
  const float aout0 = t0g ? 2.0f : 1.0f, aoff0 = t0g ? -1.0f : 0.0f;
  const float aout1 = t1g ? 2.0f : 1.0f, aoff1 = t1g ? -1.0f : 0.0f;

  // --- cell-state registers: thread owns (m = wv, j = lane*4 .. lane*4+3)
  float4 c0 = make_float4(0.f, 0.f, 0.f, 0.f);
  float4 c1 = make_float4(0.f, 0.f, 0.f, 0.f);
  const int pm = wv;
  const int pj = lane * 4;

  __syncthreads();

#pragma unroll 1
  for (int t = 0; t < Tt; ++t) {
    // ---- stage x_t into hx0[m][128..135] (f32 -> f16) ----
    if (tid < ROWS * Ff) {
      int m = tid >> 3, f = tid & 7;
      const float* xp = x + ((size_t)(bbase + m) * Tt + t) * Ff + f;
      hx0[m * S_HX0 + Hh + f] = (_Float16)(*xp);
      if (t + 1 < Tt) __builtin_prefetch(xp + Ff, 0, 1);
    }
    __syncthreads();

    // ========== layer 0: gates = [h0|x|0] @ W0cat, K0=160 (software-pipelined) ==========
    {
      v8f acc0 = {bia00, bia00, bia00, bia00, bia00, bia00, bia00, bia00};
      v8f acc1 = {bia01, bia01, bia01, bia01, bia01, bia01, bia01, bia01};
      const _Float16* ar  = hx0 + ln * S_HX0 + lh * 8;
      const _Float16* br0 = w0cat + (tn0 * 16 + ln) * K0 + lh * 8; // global (L2-hot)
      const _Float16* br1 = w0cat + (tn1 * 16 + ln) * K0 + lh * 8;
      v8h alo[2], ahi[2], b0l[2], b0h[2], b1l[2], b1h[2];
      alo[0] = *(const v8h*)(ar);
      ahi[0] = *(const v8h*)(ar + 16);
      b0l[0] = *(const v8h*)(br0); b0h[0] = *(const v8h*)(br0 + 16);
      b1l[0] = *(const v8h*)(br1); b1h[0] = *(const v8h*)(br1 + 16);
#pragma unroll
      for (int ks = 0; ks < K0 / 32; ++ks) {
        const int cur = ks & 1, nxt = cur ^ 1;
        if (ks + 1 < K0 / 32) {       // prefetch next K-step's fragments
          const int kb = (ks + 1) * 32;
          alo[nxt] = *(const v8h*)(ar + kb);
          ahi[nxt] = *(const v8h*)(ar + kb + 16);
          b0l[nxt] = *(const v8h*)(br0 + kb); b0h[nxt] = *(const v8h*)(br0 + kb + 16);
          b1l[nxt] = *(const v8h*)(br1 + kb); b1h[nxt] = *(const v8h*)(br1 + kb + 16);
        }
        v16h a = cat8(alo[cur], ahi[cur]);
        acc0 = wmma_f16(a, cat8(b0l[cur], b0h[cur]), acc0);
        acc1 = wmma_f16(a, cat8(b1l[cur], b1h[cur]), acc1);
      }
      // branchless activations scattered to gts[m][n]
#pragma unroll
      for (int r = 0; r < 8; ++r) {
        float v0 = __builtin_fmaf(aout0, fast_sigm_pre(ain0 * acc0[r]), aoff0);
        float v1 = __builtin_fmaf(aout1, fast_sigm_pre(ain1 * acc1[r]), aoff1);
        gts[(r + 8 * lh) * S_G + tn0 * 16 + ln] = v0;
        gts[(r + 8 * lh) * S_G + tn1 * 16 + ln] = v1;
      }
    }
    __syncthreads();

    // ---- layer 0 cell/hidden update: c0 = f*c0 + i*g ; h0 = o*tanh(c0) ----
    {
      const float* gr = gts + pm * S_G;
      float4 iv = *(const float4*)(gr + pj);
      float4 fv = *(const float4*)(gr + pj + 128);
      float4 gv = *(const float4*)(gr + pj + 256);
      float4 ov = *(const float4*)(gr + pj + 384);
      c0.x = __builtin_fmaf(fv.x, c0.x, iv.x * gv.x);
      c0.y = __builtin_fmaf(fv.y, c0.y, iv.y * gv.y);
      c0.z = __builtin_fmaf(fv.z, c0.z, iv.z * gv.z);
      c0.w = __builtin_fmaf(fv.w, c0.w, iv.w * gv.w);
      float h0a = ov.x * fast_tanh(c0.x), h0b = ov.y * fast_tanh(c0.y);
      float h0c = ov.z * fast_tanh(c0.z), h0d = ov.w * fast_tanh(c0.w);
      _Float16* r0 = hx0 + pm * S_HX0 + pj;        // h0_prev for next step
      _Float16* r1 = hx1 + pm * S_HX1 + Hh + pj;   // layer-1 input this step
      r0[0] = (_Float16)h0a; r0[1] = (_Float16)h0b; r0[2] = (_Float16)h0c; r0[3] = (_Float16)h0d;
      r1[0] = (_Float16)h0a; r1[1] = (_Float16)h0b; r1[2] = (_Float16)h0c; r1[3] = (_Float16)h0d;
    }
    __syncthreads();

    // ========== layer 1: gates = [h1_prev|h0_t] @ W1cat, K1=256 (software-pipelined) ==========
    {
      v8f acc0 = {bia10, bia10, bia10, bia10, bia10, bia10, bia10, bia10};
      v8f acc1 = {bia11, bia11, bia11, bia11, bia11, bia11, bia11, bia11};
      const _Float16* ar  = hx1 + ln * S_HX1 + lh * 8;
      const _Float16* br0 = W1 + (tn0 * 16 + ln) * S_W1 + lh * 8;  // LDS-resident
      const _Float16* br1 = W1 + (tn1 * 16 + ln) * S_W1 + lh * 8;
      v8h alo[2], ahi[2], b0l[2], b0h[2], b1l[2], b1h[2];
      alo[0] = *(const v8h*)(ar);
      ahi[0] = *(const v8h*)(ar + 16);
      b0l[0] = *(const v8h*)(br0); b0h[0] = *(const v8h*)(br0 + 16);
      b1l[0] = *(const v8h*)(br1); b1h[0] = *(const v8h*)(br1 + 16);
#pragma unroll
      for (int ks = 0; ks < K1 / 32; ++ks) {
        const int cur = ks & 1, nxt = cur ^ 1;
        if (ks + 1 < K1 / 32) {
          const int kb = (ks + 1) * 32;
          alo[nxt] = *(const v8h*)(ar + kb);
          ahi[nxt] = *(const v8h*)(ar + kb + 16);
          b0l[nxt] = *(const v8h*)(br0 + kb); b0h[nxt] = *(const v8h*)(br0 + kb + 16);
          b1l[nxt] = *(const v8h*)(br1 + kb); b1h[nxt] = *(const v8h*)(br1 + kb + 16);
        }
        v16h a = cat8(alo[cur], ahi[cur]);
        acc0 = wmma_f16(a, cat8(b0l[cur], b0h[cur]), acc0);
        acc1 = wmma_f16(a, cat8(b1l[cur], b1h[cur]), acc1);
      }
#pragma unroll
      for (int r = 0; r < 8; ++r) {
        float v0 = __builtin_fmaf(aout0, fast_sigm_pre(ain0 * acc0[r]), aoff0);
        float v1 = __builtin_fmaf(aout1, fast_sigm_pre(ain1 * acc1[r]), aoff1);
        gts[(r + 8 * lh) * S_G + tn0 * 16 + ln] = v0;
        gts[(r + 8 * lh) * S_G + tn1 * 16 + ln] = v1;
      }
    }
    __syncthreads();

    // ---- layer 1 cell/hidden update ----
    {
      const float* gr = gts + pm * S_G;
      float4 iv = *(const float4*)(gr + pj);
      float4 fv = *(const float4*)(gr + pj + 128);
      float4 gv = *(const float4*)(gr + pj + 256);
      float4 ov = *(const float4*)(gr + pj + 384);
      c1.x = __builtin_fmaf(fv.x, c1.x, iv.x * gv.x);
      c1.y = __builtin_fmaf(fv.y, c1.y, iv.y * gv.y);
      c1.z = __builtin_fmaf(fv.z, c1.z, iv.z * gv.z);
      c1.w = __builtin_fmaf(fv.w, c1.w, iv.w * gv.w);
      _Float16* r1 = hx1 + pm * S_HX1 + pj;        // h1_prev for next step
      r1[0] = (_Float16)(ov.x * fast_tanh(c1.x));
      r1[1] = (_Float16)(ov.y * fast_tanh(c1.y));
      r1[2] = (_Float16)(ov.z * fast_tanh(c1.z));
      r1[3] = (_Float16)(ov.w * fast_tanh(c1.w));
    }
    __syncthreads();

    // ---- output head: out[b,t,f] = h1 . w_lin[f,:] + b_lin[f] ----
    if (tid < ROWS * Ff) {
      int m = tid >> 3, fo = tid & 7;
      const _Float16* hr = hx1 + m * S_HX1;
      float s = b_lin[fo];
#pragma unroll 8
      for (int k = 0; k < Hh; ++k) s += (float)hr[k] * wl[fo * Hh + k];
      out[((size_t)(bbase + m) * Tt + t) * Ff + fo] = s;
    }
    // next iteration's x-stage only touches hx0[.,128..135] and is ordered
    // ahead of all readers by the barrier at the top of the loop body.
  }
}

extern "C" void kernel_launch(void* const* d_in, const int* in_sizes, int n_in,
                              void* d_out, int out_size, void* d_ws, size_t ws_size,
                              hipStream_t stream) {
  (void)in_sizes; (void)n_in; (void)out_size; (void)ws_size;
  const float* x     = (const float*)d_in[0];
  const float* w_ih0 = (const float*)d_in[1];
  const float* w_hh0 = (const float*)d_in[2];
  const float* b_ih0 = (const float*)d_in[3];
  const float* b_hh0 = (const float*)d_in[4];
  const float* w_ih1 = (const float*)d_in[5];
  const float* w_hh1 = (const float*)d_in[6];
  const float* b_ih1 = (const float*)d_in[7];
  const float* b_hh1 = (const float*)d_in[8];
  const float* w_lin = (const float*)d_in[9];
  const float* b_lin = (const float*)d_in[10];
  float* out = (float*)d_out;

  _Float16* w0cat = (_Float16*)d_ws;   // 512*160 halves = 160 KB

  // Build f16 layer-0 concat weights in workspace
  {
    int total = NG * K0;
    int threads = 256;
    int blocks = (total + threads - 1) / threads;
    lstm_w0cat_build<<<blocks, threads, 0, stream>>>(w_hh0, w_ih0, w0cat);
  }
  // Fused persistent scan: 4 workgroups (16 batch rows each), 512 threads, 321280 B LDS
  lstm_fused_scan<<<dim3(Bb / ROWS), dim3(512), SMEM_TOTAL, stream>>>(
      x, b_ih0, b_hh0, w_hh1, w_ih1, b_ih1, b_hh1, w_lin, b_lin, w0cat, out);
}
